// LLFQVAE_V4_21895743275555
// MI455X (gfx1250) — matrix operand order repaired and checked by reading, not compiled, not run on hardware
//
#include <hip/hip_runtime.h>
#include <math.h>

#define NB 32768
#define FD 256
#define HD 128
#define LT 128
#define NC 8192

typedef __attribute__((ext_vector_type(16))) __bf16 v16bf;
typedef __attribute__((ext_vector_type(8)))  __bf16 v8bf;
typedef __attribute__((ext_vector_type(8)))  float  v8f;
typedef int v4i __attribute__((vector_size(16)));
typedef __attribute__((address_space(1))) v4i* gptr_v4i;
typedef __attribute__((address_space(3))) v4i* lptr_v4i;

#if defined(__has_builtin)
#if __has_builtin(__builtin_amdgcn_global_load_async_to_lds_b128)
#define USE_ASYNC_LDS 1
#endif
#if __has_builtin(__builtin_amdgcn_s_wait_asynccnt)
#define HAVE_WAIT_ASYNC 1
#endif
#endif

__device__ __forceinline__ float gelu_f(float x) {
  return 0.5f * x * (1.0f + erff(x * 0.7071067811865476f));
}

__device__ __forceinline__ void wait_async_all() {
#if defined(USE_ASYNC_LDS)
#if defined(HAVE_WAIT_ASYNC)
  __builtin_amdgcn_s_wait_asynccnt(0);
#else
  asm volatile("s_wait_asynccnt 0x0" ::: "memory");
#endif
#endif
}

// ---------------- prep: Lipschitz-normalized lip weights ----------------
__global__ void k_prep_lip(const float* __restrict__ W, const float* __restrict__ ci,
                           float* __restrict__ Wp) {
  const int r = threadIdx.x;  // 128 rows
  float a = 0.f;
  for (int k = 0; k < HD; ++k) a += fabsf(W[r * HD + k]);
  float sp = log1pf(expf(ci[r]));          // softplus
  float s = fminf(1.0f, sp / a);
  for (int k = 0; k < HD; ++k) Wp[r * HD + k] = W[r * HD + k] * s;
}

// ---------------- prep: codebook -> bf16, half squared norms ----------------
__global__ void k_prep_cb(const float* __restrict__ cb, __bf16* __restrict__ cbb,
                          float* __restrict__ cn) {
  const int j = blockIdx.x * 256 + threadIdx.x;  // 8192 rows
  float s = 0.f;
  for (int k = 0; k < LT; ++k) {
    float v = cb[(size_t)j * LT + k];
    s += v * v;
    cbb[(size_t)j * LT + k] = (__bf16)v;
  }
  cn[j] = 0.5f * s;
}

// ---------------- fused encoder: x -> z_e (f32 + bf16) ----------------
__global__ __launch_bounds__(64) void k_encoder(
    const float* __restrict__ x,
    const float* __restrict__ ew1, const float* __restrict__ eb1,
    const float* __restrict__ ew2, const float* __restrict__ eb2,
    const float* __restrict__ Wp,  const float* __restrict__ lipb,
    float* __restrict__ zef, __bf16* __restrict__ zeb) {
  __shared__ float h1s[64 * 65];
  __shared__ float h2s[64 * 129];
  const int t = threadIdx.x;
  const size_t row = (size_t)blockIdx.x * 64 + t;
  const float* xr = x + row * FD;

  // layer1: 256 -> 64, gelu
  for (int jc = 0; jc < 4; ++jc) {
    float s[16];
#pragma unroll
    for (int jj = 0; jj < 16; ++jj) s[jj] = eb1[jc * 16 + jj];
    for (int kc = 0; kc < 8; ++kc) {
      float xv[32];
#pragma unroll
      for (int u = 0; u < 32; ++u) xv[u] = xr[kc * 32 + u];
#pragma unroll
      for (int jj = 0; jj < 16; ++jj) {
        const float* wrow = ew1 + (size_t)(jc * 16 + jj) * FD + kc * 32;
#pragma unroll
        for (int u = 0; u < 32; ++u) s[jj] += xv[u] * wrow[u];
      }
    }
#pragma unroll
    for (int jj = 0; jj < 16; ++jj) h1s[t * 65 + jc * 16 + jj] = gelu_f(s[jj]);
  }
  // layer2: 64 -> 128, gelu
  for (int jc = 0; jc < 8; ++jc) {
    float s[16];
#pragma unroll
    for (int jj = 0; jj < 16; ++jj) s[jj] = eb2[jc * 16 + jj];
    for (int kc = 0; kc < 4; ++kc) {
      float hv[16];
#pragma unroll
      for (int u = 0; u < 16; ++u) hv[u] = h1s[t * 65 + kc * 16 + u];
#pragma unroll
      for (int jj = 0; jj < 16; ++jj) {
        const float* wrow = ew2 + (size_t)(jc * 16 + jj) * 64 + kc * 16;
#pragma unroll
        for (int u = 0; u < 16; ++u) s[jj] += hv[u] * wrow[u];
      }
    }
#pragma unroll
    for (int jj = 0; jj < 16; ++jj) h2s[t * 129 + jc * 16 + jj] = gelu_f(s[jj]);
  }
  // lip: 128 -> 128, sigmoid
  for (int jc = 0; jc < 8; ++jc) {
    float s[16];
#pragma unroll
    for (int jj = 0; jj < 16; ++jj) s[jj] = lipb[jc * 16 + jj];
    for (int kc = 0; kc < 8; ++kc) {
      float hv[16];
#pragma unroll
      for (int u = 0; u < 16; ++u) hv[u] = h2s[t * 129 + kc * 16 + u];
#pragma unroll
      for (int jj = 0; jj < 16; ++jj) {
        const float* wrow = Wp + (size_t)(jc * 16 + jj) * HD + kc * 16;
#pragma unroll
        for (int u = 0; u < 16; ++u) s[jj] += hv[u] * wrow[u];
      }
    }
#pragma unroll
    for (int jj = 0; jj < 16; ++jj) {
      float z = 1.0f / (1.0f + expf(-s[jj]));
      zef[row * LT + jc * 16 + jj] = z;
      zeb[row * LT + jc * 16 + jj] = (__bf16)z;
    }
  }
}

// ---------------- LFQ quantizer ----------------
// Block = 128 threads = 4 waves; each wave owns 32 rows (two A tiles) so every
// B fragment feeds two WMMAs (halves LDS read BW per WMMA, two independent
// accumulate chains). Codebook streamed 64 codes/step through double-buffered
// LDS via async global->LDS copies (ASYNCcnt) overlapped with WMMA compute.
// B fragments are software-pipelined one step ahead in registers so the LDS
// loads of step i+1 are in flight during the WMMAs of step i.
#define QTILE (64 * 136)

__device__ __forceinline__ v16bf load_bfrag(const __bf16* buf, int s4, int ks,
                                            int l16, int half) {
  // B 32x16 layout: lane holds col N=l16, K {8h..8h+7, 16+8h..16+8h+7}, h=lane/16
  const __bf16* bp = buf + (s4 * 16 + l16) * 136 + ks * 32 + half * 8;
  v8bf blo = *(const v8bf*)(bp);
  v8bf bhi = *(const v8bf*)(bp + 16);
  v16bf B;
#pragma unroll
  for (int u = 0; u < 8; ++u) { B[u] = blo[u]; B[u + 8] = bhi[u]; }
  return B;
}

__global__ __launch_bounds__(128) void k_quant(
    const __bf16* __restrict__ ze, const __bf16* __restrict__ cbb,
    const float* __restrict__ cn, int* __restrict__ oidx) {
  __shared__ __attribute__((aligned(16))) __bf16 tile[2 * QTILE];
  const int t = threadIdx.x, lane = t & 31, wave = t >> 5;
  const int half = lane >> 4, l16 = lane & 15;
  const int rowBase = blockIdx.x * 128 + wave * 32;

  // A fragments: 2 tiles x 16 rows x 128 dims of z_e, resident in registers.
  // ISA 16-bit A 16x32 layout: lanes 0-15 hold K {0..7,16..23}, lanes 16-31 hold K {8..15,24..31}.
  v16bf A0[4], A1[4];
  {
    const __bf16* zr0 = ze + (size_t)(rowBase + l16) * LT;
    const __bf16* zr1 = ze + (size_t)(rowBase + 16 + l16) * LT;
#pragma unroll
    for (int ks = 0; ks < 4; ++ks) {
      v8bf lo0 = *(const v8bf*)(zr0 + ks * 32 + half * 8);
      v8bf hi0 = *(const v8bf*)(zr0 + ks * 32 + 16 + half * 8);
      v8bf lo1 = *(const v8bf*)(zr1 + ks * 32 + half * 8);
      v8bf hi1 = *(const v8bf*)(zr1 + ks * 32 + 16 + half * 8);
#pragma unroll
      for (int u = 0; u < 8; ++u) {
        A0[ks][u] = lo0[u]; A0[ks][u + 8] = hi0[u];
        A1[ks][u] = lo1[u]; A1[ks][u + 8] = hi1[u];
      }
    }
  }

  float best0[8], best1[8];
  int bidx0[8], bidx1[8];
#pragma unroll
  for (int v = 0; v < 8; ++v) {
    best0[v] = -3.4e38f; bidx0[v] = 0;
    best1[v] = -3.4e38f; bidx1[v] = 0;
  }

  // stage 64 codes x 128 bf16 = 1024 x 16B chunks; 8 chunks per thread
  auto stage = [&](__bf16* dst, int j0) {
#pragma unroll
    for (int q = 0; q < 8; ++q) {
      int ch = q * 128 + t;
      int r = ch >> 4, c = ch & 15;
#if defined(USE_ASYNC_LDS)
      __builtin_amdgcn_global_load_async_to_lds_b128(
          (gptr_v4i)(cbb + (size_t)(j0 + r) * LT + c * 8),
          (lptr_v4i)(dst + r * 136 + c * 8), 0, 0);
#else
      *(v8bf*)(dst + r * 136 + c * 8) =
          *(const v8bf*)(cbb + (size_t)(j0 + r) * LT + c * 8);
#endif
    }
  };

  const int NSTEP = NC / 64;  // 128
  stage(tile, 0);
  wait_async_all();
  __syncthreads();

  for (int jt = 0; jt < NSTEP; ++jt) {
    const __bf16* buf = tile + (jt & 1) * QTILE;
    if (jt + 1 < NSTEP) stage(tile + ((jt + 1) & 1) * QTILE, (jt + 1) * 64);

    v16bf Bcur = load_bfrag(buf, 0, 0, l16, half);
#pragma unroll
    for (int s4 = 0; s4 < 4; ++s4) {
      const int jb = jt * 64 + s4 * 16;
      const float cnv = cn[jb + l16];
      v8f a0 = {}, a1 = {};
#pragma unroll
      for (int ks = 0; ks < 4; ++ks) {
        const int ni = s4 * 4 + ks + 1;              // next flattened step
        v16bf Bnext = (ni < 16) ? load_bfrag(buf, ni >> 2, ni & 3, l16, half)
                                : Bcur;              // no prefetch past barrier
        a0 = __builtin_amdgcn_wmma_f32_16x16x32_bf16(
            false, A0[ks], false, Bcur, (short)0, a0, false, false);
        a1 = __builtin_amdgcn_wmma_f32_16x16x32_bf16(
            false, A1[ks], false, Bcur, (short)0, a1, false, false);
        Bcur = Bnext;
      }
#pragma unroll
      for (int v = 0; v < 8; ++v) {
        float s0 = a0[v] - cnv;  // argmax(z.c - 0.5||c||^2) == argmin distance
        if (s0 > best0[v]) { best0[v] = s0; bidx0[v] = jb + l16; }
        float s1 = a1[v] - cnv;
        if (s1 > best1[v]) { best1[v] = s1; bidx1[v] = jb + l16; }
      }
    }
    wait_async_all();
    __syncthreads();
  }

  // reduce over the 16 N-lanes within each half-wave; first-index tie-break
#pragma unroll
  for (int off = 1; off < 16; off <<= 1) {
#pragma unroll
    for (int v = 0; v < 8; ++v) {
      float ob0 = __shfl_xor(best0[v], off, 32);
      int   oi0 = __shfl_xor(bidx0[v], off, 32);
      if (ob0 > best0[v] || (ob0 == best0[v] && oi0 < bidx0[v])) {
        best0[v] = ob0; bidx0[v] = oi0;
      }
      float ob1 = __shfl_xor(best1[v], off, 32);
      int   oi1 = __shfl_xor(bidx1[v], off, 32);
      if (ob1 > best1[v] || (ob1 == best1[v] && oi1 < bidx1[v])) {
        best1[v] = ob1; bidx1[v] = oi1;
      }
    }
  }
  if (l16 == 0) {
#pragma unroll
    for (int v = 0; v < 8; ++v) {
      oidx[rowBase + half * 8 + v] = bidx0[v];        // C-layout: M = v + 8*half
      oidx[rowBase + 16 + half * 8 + v] = bidx1[v];
    }
  }
}

// ---------------- gather + fused decoder + loss partials ----------------
__global__ __launch_bounds__(64) void k_decoder(
    const int* __restrict__ idxs, const float* __restrict__ cb,
    const float* __restrict__ zef, const float* __restrict__ x,
    const float* __restrict__ dw1, const float* __restrict__ db1,
    const float* __restrict__ dw2, const float* __restrict__ db2,
    const float* __restrict__ ow,  const float* __restrict__ obias,
    float* __restrict__ outz, float* __restrict__ part1, float* __restrict__ part2) {
  __shared__ float As[64 * 129];
  __shared__ float Bs[64 * 65];
  __shared__ float red[64];
  const int t = threadIdx.x;
  const size_t row = (size_t)blockIdx.x * 64 + t;
  const int code = idxs[row];
  const float* zq = cb + (size_t)code * LT;

  float e2 = 0.f;
  for (int k = 0; k < LT; ++k) {
    float zv = zq[k];
    As[t * 129 + k] = zv;
    outz[row * LT + k] = zv;                 // z_latent output
    float d = zv - zef[row * LT + k];
    e2 += d * d;                              // ||z_q - z_e||^2
  }
  // dec1: 128 -> 64, gelu
  for (int jc = 0; jc < 4; ++jc) {
    float s[16];
#pragma unroll
    for (int jj = 0; jj < 16; ++jj) s[jj] = db1[jc * 16 + jj];
    for (int kc = 0; kc < 8; ++kc) {
      float hv[16];
#pragma unroll
      for (int u = 0; u < 16; ++u) hv[u] = As[t * 129 + kc * 16 + u];
#pragma unroll
      for (int jj = 0; jj < 16; ++jj) {
        const float* wrow = dw1 + (size_t)(jc * 16 + jj) * HD + kc * 16;
#pragma unroll
        for (int u = 0; u < 16; ++u) s[jj] += hv[u] * wrow[u];
      }
    }
#pragma unroll
    for (int jj = 0; jj < 16; ++jj) Bs[t * 65 + jc * 16 + jj] = gelu_f(s[jj]);
  }
  // dec2: 64 -> 128, gelu (overwrite As, z_q dead)
  for (int jc = 0; jc < 8; ++jc) {
    float s[16];
#pragma unroll
    for (int jj = 0; jj < 16; ++jj) s[jj] = db2[jc * 16 + jj];
    for (int kc = 0; kc < 4; ++kc) {
      float hv[16];
#pragma unroll
      for (int u = 0; u < 16; ++u) hv[u] = Bs[t * 65 + kc * 16 + u];
#pragma unroll
      for (int jj = 0; jj < 16; ++jj) {
        const float* wrow = dw2 + (size_t)(jc * 16 + jj) * 64 + kc * 16;
#pragma unroll
        for (int u = 0; u < 16; ++u) s[jj] += hv[u] * wrow[u];
      }
    }
#pragma unroll
    for (int jj = 0; jj < 16; ++jj) As[t * 129 + jc * 16 + jj] = gelu_f(s[jj]);
  }
  // out: 128 -> 256; accumulate ||x_recon - x||^2 (x_recon never stored)
  const float* xr = x + row * FD;
  float e1 = 0.f;
  for (int jc = 0; jc < 16; ++jc) {
    float s[16];
#pragma unroll
    for (int jj = 0; jj < 16; ++jj) s[jj] = obias[jc * 16 + jj];
    for (int kc = 0; kc < 8; ++kc) {
      float hv[16];
#pragma unroll
      for (int u = 0; u < 16; ++u) hv[u] = As[t * 129 + kc * 16 + u];
#pragma unroll
      for (int jj = 0; jj < 16; ++jj) {
        const float* wrow = ow + (size_t)(jc * 16 + jj) * HD + kc * 16;
#pragma unroll
        for (int u = 0; u < 16; ++u) s[jj] += hv[u] * wrow[u];
      }
    }
#pragma unroll
    for (int jj = 0; jj < 16; ++jj) {
      float d = s[jj] - xr[jc * 16 + jj];
      e1 += d * d;
    }
  }
  // deterministic fixed-order block reductions
  red[t] = e1; __syncthreads();
  for (int sft = 32; sft > 0; sft >>= 1) { if (t < sft) red[t] += red[t + sft]; __syncthreads(); }
  if (t == 0) part1[blockIdx.x] = red[0];
  __syncthreads();
  red[t] = e2; __syncthreads();
  for (int sft = 32; sft > 0; sft >>= 1) { if (t < sft) red[t] += red[t + sft]; __syncthreads(); }
  if (t == 0) part2[blockIdx.x] = red[0];
}

// ---------------- final deterministic loss ----------------
__global__ void k_final(const float* __restrict__ p1, const float* __restrict__ p2,
                        float* __restrict__ loss) {
  if (threadIdx.x == 0 && blockIdx.x == 0) {
    float a = 0.f, b = 0.f;
    for (int i = 0; i < NB / 64; ++i) { a += p1[i]; b += p2[i]; }
    *loss = a / (float)((size_t)NB * FD) + 0.5f * b / (float)((size_t)NB * LT);
  }
}

extern "C" void kernel_launch(void* const* d_in, const int* in_sizes, int n_in,
                              void* d_out, int out_size, void* d_ws, size_t ws_size,
                              hipStream_t stream) {
  (void)in_sizes; (void)n_in; (void)out_size; (void)ws_size;
  const float* x     = (const float*)d_in[0];
  const float* ew1   = (const float*)d_in[1];
  const float* eb1   = (const float*)d_in[2];
  const float* ew2   = (const float*)d_in[3];
  const float* eb2   = (const float*)d_in[4];
  const float* lipW  = (const float*)d_in[5];
  const float* lipb  = (const float*)d_in[6];
  const float* lipci = (const float*)d_in[7];
  const float* cb    = (const float*)d_in[8];
  const float* dw1   = (const float*)d_in[9];
  const float* db1   = (const float*)d_in[10];
  const float* dw2   = (const float*)d_in[11];
  const float* db2   = (const float*)d_in[12];
  const float* ow    = (const float*)d_in[13];
  const float* ob    = (const float*)d_in[14];
  float* out = (float*)d_out;

  char* w = (char*)d_ws;
  float*  zef  = (float*)(w);                                // 16 MB
  __bf16* zeb  = (__bf16*)(w + 16777216);                    //  8 MB
  __bf16* cbb  = (__bf16*)(w + 25165824);                    //  2 MB
  float*  cn   = (float*)(w + 27262976);                     // 32 KB
  float*  Wp   = (float*)(w + 27295744);                     // 64 KB
  int*    idxs = (int*)(w + 27361280);                       // 128 KB
  float*  p1   = (float*)(w + 27492352);                     // 2 KB
  float*  p2   = (float*)(w + 27494400);                     // 2 KB

  k_prep_lip<<<1, 128, 0, stream>>>(lipW, lipci, Wp);
  k_prep_cb<<<NC / 256, 256, 0, stream>>>(cb, cbb, cn);
  k_encoder<<<NB / 64, 64, 0, stream>>>(x, ew1, eb1, ew2, eb2, Wp, lipb, zef, zeb);
  k_quant<<<NB / 128, 128, 0, stream>>>(zeb, cbb, cn, idxs);
  k_decoder<<<NB / 64, 64, 0, stream>>>(idxs, cb, zef, x, dw1, db1, dw2, db2,
                                        ow, ob, out, p1, p2);
  k_final<<<1, 1, 0, stream>>>(p1, p2, out + (size_t)NB * LT);
}
